// CombinedElevationEncoder_17789754540233
// MI455X (gfx1250) — matrix operand order repaired and checked by reading, not compiled, not run on hardware
//
#include <hip/hip_runtime.h>
#include <math.h>

// ---------------------------------------------------------------------------
// CDNA5 (gfx1250) wave32 WMMA types and helpers
// ---------------------------------------------------------------------------
typedef __attribute__((ext_vector_type(16))) __bf16 v16bf;
typedef __attribute__((ext_vector_type(8)))  float  v8f;

union ABf  { v16bf v; __bf16 e[16]; unsigned u[8]; uint4 q[2]; };
union AccF { v8f   v; float  e[8];  };
union St8  { uint4 q; __bf16 e[8];  };

// native f32 -> bf16 convert (RNE); lowers to v_cvt_*bf16_f32 on gfx1250
__device__ __forceinline__ __bf16 f2bf(float x) { return (__bf16)x; }

// 16-bit A-matrix 16x32 lane layout (contiguous, same K order as B):
// lanes 0-15: K=0..15 ; lanes 16-31: K=16..31
__device__ __forceinline__ int a16_kidx(int lane, int i) {
  return ((lane & 16) ? 16 : 0) + i;
}
// 16-bit B-matrix 32x16 lane layout: lanes 0-15 hold K=0..15 across the 8
// VGPRs (contiguous), lanes 16-31 hold K=16..31.
__device__ __forceinline__ int b16_kbase(int lane, int ks) {
  return ks * 32 + ((lane & 16) ? 16 : 0);
}

__device__ __forceinline__ float sigm(float x) { return 1.f / (1.f + __expf(-x)); }

// ---------------------------------------------------------------------------
// Kernel Z: zero the pooled-sum accumulator (graph-safe re-init each call)
// ---------------------------------------------------------------------------
__global__ void k_zero(float* __restrict__ p, int n) {
  for (int i = blockIdx.x * 256 + threadIdx.x; i < n; i += gridDim.x * 256) p[i] = 0.f;
}

// ---------------------------------------------------------------------------
// Kernel A: the three input GEMMs  relu(X[32,256] @ W[256,512] + b)
// 192 waves total, one 16x16 output tile per wave, 8 k-steps of bf16 WMMA.
// ---------------------------------------------------------------------------
__global__ __launch_bounds__(256) void k_gemm3(
    const float* __restrict__ dist, const float* __restrict__ az, const float* __restrict__ el,
    const float* __restrict__ Wd, const float* __restrict__ bd,
    const float* __restrict__ Wa, const float* __restrict__ ba,
    const float* __restrict__ We, const float* __restrict__ be,
    float* __restrict__ out, float* __restrict__ baseWs) {
  int wid  = blockIdx.x * 8 + (threadIdx.x >> 5);
  int lane = threadIdx.x & 31;
  int mat  = wid >> 6;        // 0=dist 1=az 2=elevation
  int tile = wid & 63;
  int mt = tile & 1, nt = tile >> 1;

  const float *X, *W, *bias;
  if (mat == 0)      { X = dist; W = Wd; bias = bd; }
  else if (mat == 1) { X = az;   W = Wa; bias = ba; }
  else               { X = el;   W = We; bias = be; }

  AccF acc;
#pragma unroll
  for (int r = 0; r < 8; ++r) acc.e[r] = 0.f;

  int mrow = mt * 16 + (lane & 15);
  int ncol = nt * 16 + (lane & 15);

  for (int ks = 0; ks < 8; ++ks) {
    ABf a, b;
#pragma unroll
    for (int j = 0; j < 8; ++j) {         // A K-pairs are memory-adjacent
      int ka = ks * 32 + a16_kidx(lane, 2 * j);
      float2 xv = *(const float2*)&X[mrow * 256 + ka];
      a.e[2 * j]     = f2bf(xv.x);
      a.e[2 * j + 1] = f2bf(xv.y);
    }
    int kb = b16_kbase(lane, ks);
#pragma unroll
    for (int i = 0; i < 16; ++i) b.e[i] = f2bf(W[(kb + i) * 512 + ncol]);
    acc.v = __builtin_amdgcn_wmma_f32_16x16x32_bf16(false, a.v, false, b.v,
                                                    (short)0, acc.v, false, false);
  }
#pragma unroll
  for (int r = 0; r < 8; ++r) {
    int m = mt * 16 + ((lane < 16) ? r : r + 8);
    float v = fmaxf(0.f, acc.e[r] + bias[ncol]);
    if (mat == 2) baseWs[m * 512 + ncol] = v;           // relu'd base
    else          out[m * 1536 + mat * 512 + ncol] = v; // dist / az outputs
  }
}

// ---------------------------------------------------------------------------
// Kernel B: fused conv1(2->8,5x7) + relu + conv2(8->8,3x5) + relu + 4x4 pool.
// One WG per (batch, ph, 64-wide W subtile). LDS tiles are CHANNEL-LAST and
// the implicit-GEMM K order is (kh, kw, ic) so each lane's 16 contiguous
// B-matrix K halves are contiguous LDS bytes: conv2 B = 2x ds_load_b128 per
// k-step, conv1 B = 8x ds_load_b32, conv1 store = 1x ds_store_b128.
// ---------------------------------------------------------------------------
#define WT 64
__global__ __launch_bounds__(256) void k_conv(
    const float* __restrict__ rs, const float* __restrict__ w1, const float* __restrict__ b1,
    const float* __restrict__ w2, const float* __restrict__ b2,
    float* __restrict__ pooledSum) {
  // input tile, channel-last: [38 rows][80 cols][2 ch]  (rows h0-3.., cols w0-5..)
  __shared__ __align__(16) __bf16 inT2[38 * 80 * 2];
  // conv1 output, channel-last: [34 rows][80 cols][8 ch] (rows h0-1.., cols w0-2..)
  __shared__ __align__(16) __bf16 em1c[34 * 80 * 8];
  __shared__ float chSum[8];

  int b = blockIdx.x, ph = blockIdx.y, wb = blockIdx.z;
  int h0 = ph * 32, w0 = wb * WT;
  int pw = wb / 6;                       // 6 * 64 = 384 = pooled W block
  int tid = threadIdx.x, lane = tid & 31, warp = tid >> 5;

  if (tid < 8) chSum[tid] = 0.f;

  // phase 1: stage both channels of each input pixel as one packed pair
  for (int idx = tid; idx < 38 * 74; idx += 256) {
    int tr = idx / 74, tc = idx % 74;
    int gh = h0 - 3 + tr, gw = w0 - 5 + tc;
    float v0 = 0.f, v1 = 0.f;
    if (gh >= 0 && gh < 128 && gw >= 0 && gw < 1536) {
      const float* p = rs + ((size_t)(b * 2) * 128 + gh) * 1536 + gw;
      v0 = p[0];
      v1 = p[128 * 1536];
      __builtin_prefetch(p + 1536, 0, 1);   // global_prefetch_b8 of next row
    }
    St8 st;
    st.e[0] = f2bf(v0); st.e[1] = f2bf(v1);
    *(unsigned*)&inT2[(tr * 80 + tc) * 2] = st.q.x;
  }
  __syncthreads();

  // conv1 A-matrix (weights 8x70 padded to 16x96); K order kk=(kh*7+kw)*2+ic
  ABf a1[3];
  {
    int m = lane & 15;
#pragma unroll
    for (int ks = 0; ks < 3; ++ks)
#pragma unroll
      for (int i = 0; i < 16; ++i) {
        int kk = ks * 32 + a16_kidx(lane, i);
        float wv = 0.f;
        if (m < 8 && kk < 70) {
          int p = kk >> 1, ic = kk & 1;
          int kh = p / 7, kw = p % 7;
          wv = w1[m * 70 + ic * 35 + kh * 7 + kw];
        }
        a1[ks].e[i] = f2bf(wv);
      }
  }
  // phase 2: conv1 -> em1c (34 rows x 68 valid cols, 5 chunks of 16)
  for (int ti = warp; ti < 34 * 5; ti += 8) {
    int trow = ti / 5, chunk = ti % 5;
    int n = lane & 15;
    AccF acc;
#pragma unroll
    for (int r = 0; r < 8; ++r) acc.e[r] = 0.f;
#pragma unroll
    for (int ks = 0; ks < 3; ++ks) {
      ABf bm;
      int pbase = (b16_kbase(lane, ks) >> 1);     // first (kh,kw) tap index
#pragma unroll
      for (int j = 0; j < 8; ++j) {               // 8 taps x 2 channels
        int p = pbase + j;
        unsigned pk = 0u;
        if (p < 35) {
          int kh = p / 7, kw = p % 7;
          int tcc = chunk * 16 + n + kw;
          if (tcc < 74) pk = *(const unsigned*)&inT2[((trow + kh) * 80 + tcc) * 2];
        }
        bm.u[j] = pk;
      }
      acc.v = __builtin_amdgcn_wmma_f32_16x16x32_bf16(false, a1[ks].v, false, bm.v,
                                                      (short)0, acc.v, false, false);
    }
    if (lane < 16) {                       // rows 0..7 = the 8 real channels
      int tc = chunk * 16 + lane;
      if (tc < WT + 4) {
        St8 st;
#pragma unroll
        for (int r = 0; r < 8; ++r) st.e[r] = f2bf(fmaxf(0.f, acc.e[r] + b1[r]));
        *(uint4*)&em1c[(trow * 80 + tc) * 8] = st.q;   // one ds_store_b128
      }
    }
  }
  __syncthreads();

  // conv2 A-matrix (weights 8x120 padded to 16x128); K order kk=(kh*5+kw)*8+ic
  ABf a2[4];
  {
    int m = lane & 15;
#pragma unroll
    for (int ks = 0; ks < 4; ++ks)
#pragma unroll
      for (int i = 0; i < 16; ++i) {
        int kk = ks * 32 + a16_kidx(lane, i);
        float wv = 0.f;
        if (m < 8 && kk < 120) {
          int p = kk >> 3, ic = kk & 7;
          int kh = p / 5, kw = p % 5;
          wv = w2[m * 120 + ic * 15 + kh * 5 + kw];
        }
        a2[ks].e[i] = f2bf(wv);
      }
  }
  // phase 3: conv2 + relu + per-channel register accumulation over 32x64
  float regSum[8];
#pragma unroll
  for (int r = 0; r < 8; ++r) regSum[r] = 0.f;

  for (int ti = warp; ti < 32 * (WT / 16); ti += 8) {
    int r = ti / (WT / 16), chunk = ti % (WT / 16);
    int n = lane & 15;
    AccF acc;
#pragma unroll
    for (int q = 0; q < 8; ++q) acc.e[q] = 0.f;
#pragma unroll
    for (int ks = 0; ks < 4; ++ks) {
      ABf bm;
      int pbase = (b16_kbase(lane, ks) >> 3);     // first (kh,kw) tap index
#pragma unroll
      for (int g = 0; g < 2; ++g) {               // 2 taps x 8 channels
        int p = pbase + g;
        if (p < 15) {
          int kh = p / 5, kw = p % 5;
          int tcc = chunk * 16 + n + kw;
          bm.q[g] = *(const uint4*)&em1c[((r + kh) * 80 + tcc) * 8]; // ds_load_b128
        } else {
          bm.q[g] = make_uint4(0u, 0u, 0u, 0u);
        }
      }
      acc.v = __builtin_amdgcn_wmma_f32_16x16x32_bf16(false, a2[ks].v, false, bm.v,
                                                      (short)0, acc.v, false, false);
    }
    if (lane < 16) {
#pragma unroll
      for (int rr = 0; rr < 8; ++rr) regSum[rr] += fmaxf(0.f, acc.e[rr] + b2[rr]);
    }
  }
  if (lane < 16) {
#pragma unroll
    for (int rr = 0; rr < 8; ++rr) atomicAdd(&chSum[rr], regSum[rr]);
  }
  __syncthreads();
  if (tid < 8)
    atomicAdd(&pooledSum[((b * 8 + tid) * 4 + ph) * 4 + pw], chSum[tid]);
}

// ---------------------------------------------------------------------------
// Kernel C: learned = pooled[32,128] @ W_res[128,512] + b_res (bf16 WMMA)
// ---------------------------------------------------------------------------
__global__ __launch_bounds__(256) void k_learned(
    const float* __restrict__ pooledSum, const float* __restrict__ Wres,
    const float* __restrict__ bres, float* __restrict__ learned) {
  int wid = blockIdx.x * 8 + (threadIdx.x >> 5);
  int lane = threadIdx.x & 31;
  int mt = wid & 1, nt = wid >> 1;
  const float inv = 1.f / 12288.f;        // pool block 32 x 384
  AccF acc;
#pragma unroll
  for (int r = 0; r < 8; ++r) acc.e[r] = 0.f;
  int mrow = mt * 16 + (lane & 15);
  int ncol = nt * 16 + (lane & 15);
  for (int ks = 0; ks < 4; ++ks) {
    ABf a, b;
#pragma unroll
    for (int j = 0; j < 8; ++j) {
      int ka = ks * 32 + a16_kidx(lane, 2 * j);
      float2 xv = *(const float2*)&pooledSum[mrow * 128 + ka];
      a.e[2 * j]     = f2bf(xv.x * inv);
      a.e[2 * j + 1] = f2bf(xv.y * inv);
    }
    int kb = b16_kbase(lane, ks);
#pragma unroll
    for (int i = 0; i < 16; ++i) b.e[i] = f2bf(Wres[(kb + i) * 512 + ncol]);
    acc.v = __builtin_amdgcn_wmma_f32_16x16x32_bf16(false, a.v, false, b.v,
                                                    (short)0, acc.v, false, false);
  }
#pragma unroll
  for (int r = 0; r < 8; ++r) {
    int m = mt * 16 + ((lane < 16) ? r : r + 8);
    learned[m * 512 + ncol] = acc.e[r] + bres[ncol];
  }
}

// ---------------------------------------------------------------------------
// Kernel D: spiking Conv2dLSTM scan. One WG per batch, thread = (h, w2),
// mem/frame mirrored in LDS for the 3-tap H halo, 128 sequential steps.
// ---------------------------------------------------------------------------
__global__ __launch_bounds__(256) void k_scan(
    const float* __restrict__ rs, const float* __restrict__ sw,
    const float* __restrict__ sb, float* __restrict__ ctx) {
  __shared__ float frameL[2][130];
  __shared__ float memL[2][4][130];
  __shared__ float swS[240];
  __shared__ float sbS[16];
  __shared__ float chAcc[4];

  int b = blockIdx.x, tid = threadIdx.x;
  int h = tid & 127, w2 = tid >> 7;
  if (tid < 240) swS[tid] = sw[tid];
  if (tid < 16)  sbS[tid] = sb[tid];
  if (tid < 4)   chAcc[tid] = 0.f;
  if (tid < 2) {
    frameL[tid][0] = 0.f; frameL[tid][129] = 0.f;
#pragma unroll
    for (int c = 0; c < 4; ++c) { memL[tid][c][0] = 0.f; memL[tid][c][129] = 0.f; }
  }
  float syn[4] = {0, 0, 0, 0}, mem[4] = {0, 0, 0, 0}, spk[4] = {0, 0, 0, 0};
  const float* rbase = rs + ((size_t)(b * 2 + w2) * 128 + h) * 1536;
  __syncthreads();

  for (int t = 0; t < 128; ++t) {
    float f = 0.f;
#pragma unroll
    for (int j = 0; j < 12; ++j) f += rbase[t * 12 + j];   // avg_pool1d k=12
    f *= (1.f / 12.f);
    frameL[w2][h + 1] = f;
#pragma unroll
    for (int c = 0; c < 4; ++c) memL[w2][c][h + 1] = mem[c];
    __syncthreads();

    float x0m = frameL[w2][h], x0p = frameL[w2][h + 2];
    float xmm[4], xmp[4];
#pragma unroll
    for (int c = 0; c < 4; ++c) { xmm[c] = memL[w2][c][h]; xmp[c] = memL[w2][c][h + 2]; }
    float g[16];
#pragma unroll
    for (int og = 0; og < 16; ++og) {
      const float* w = swS + og * 15;
      float s = sbS[og] + w[0] * x0m + w[1] * f + w[2] * x0p;
#pragma unroll
      for (int c = 0; c < 4; ++c)
        s += w[3 + c * 3] * xmm[c] + w[4 + c * 3] * mem[c] + w[5 + c * 3] * xmp[c];
      g[og] = s;
    }
#pragma unroll
    for (int c = 0; c < 4; ++c) {
      float ig = sigm(g[c]), fg = sigm(g[4 + c]);
      float gg = tanhf(g[8 + c]), og_ = sigm(g[12 + c]);
      syn[c] = fg * syn[c] + ig * gg;
      mem[c] = og_ * tanhf(syn[c]);
      spk[c] += (mem[c] - 1.f > 0.f) ? 1.f : 0.f;        // spike(mem - 1)
    }
    __syncthreads();
  }
#pragma unroll
  for (int c = 0; c < 4; ++c) atomicAdd(&chAcc[c], spk[c]);
  __syncthreads();
  if (tid < 4) ctx[b * 4 + tid] = chAcc[tid] * (1.f / 32768.f);  // mean over T*H*2
}

// ---------------------------------------------------------------------------
// Kernel E: elev = relu(base + cnn_scale*learned + sconv_scale*(ctx@W_proj+b))
// ---------------------------------------------------------------------------
__global__ __launch_bounds__(256) void k_final(
    const float* __restrict__ baseWs, const float* __restrict__ learned,
    const float* __restrict__ ctx, const float* __restrict__ Wproj,
    const float* __restrict__ bproj, const float* __restrict__ cnnGain,
    const float* __restrict__ scGain, float* __restrict__ out) {
  int i = blockIdx.x * 256 + threadIdx.x;  // 32*512
  int b = i >> 9, n = i & 511;
  float cs = 0.5f * sigm(cnnGain[0]);
  float ss = 0.4f * sigm(scGain[0]);
  float sres = bproj[n];
#pragma unroll
  for (int c = 0; c < 4; ++c) sres += ctx[b * 4 + c] * Wproj[c * 512 + n];
  out[b * 1536 + 1024 + n] = fmaxf(0.f, baseWs[i] + cs * learned[i] + ss * sres);
}

// ---------------------------------------------------------------------------
extern "C" void kernel_launch(void* const* d_in, const int* in_sizes, int n_in,
                              void* d_out, int out_size, void* d_ws, size_t ws_size,
                              hipStream_t stream) {
  const float* distance = (const float*)d_in[0];
  const float* azimuth  = (const float*)d_in[1];
  const float* elev     = (const float*)d_in[2];
  const float* rs       = (const float*)d_in[3];
  const float* W_dist   = (const float*)d_in[4];
  const float* b_dist   = (const float*)d_in[5];
  const float* W_az     = (const float*)d_in[6];
  const float* b_az     = (const float*)d_in[7];
  const float* W_el     = (const float*)d_in[8];
  const float* b_el     = (const float*)d_in[9];
  const float* conv1_w  = (const float*)d_in[10];
  const float* conv1_b  = (const float*)d_in[11];
  const float* conv2_w  = (const float*)d_in[12];
  const float* conv2_b  = (const float*)d_in[13];
  const float* W_res    = (const float*)d_in[14];
  const float* b_res    = (const float*)d_in[15];
  const float* sconv_w  = (const float*)d_in[16];
  const float* sconv_b  = (const float*)d_in[17];
  const float* W_proj   = (const float*)d_in[18];
  const float* b_proj   = (const float*)d_in[19];
  const float* cnn_gain = (const float*)d_in[20];
  const float* sc_gain  = (const float*)d_in[21];
  float* out = (float*)d_out;

  float* ws        = (float*)d_ws;
  float* baseWs    = ws;             // 32*512  = 16384 f32
  float* pooledSum = ws + 16384;     // 32*8*16 =  4096 f32
  float* learned   = ws + 20480;     // 32*512  = 16384 f32
  float* ctx       = ws + 36864;     // 32*4    =   128 f32

  k_zero<<<16, 256, 0, stream>>>(pooledSum, 4096);
  k_gemm3<<<24, 256, 0, stream>>>(distance, azimuth, elev,
                                  W_dist, b_dist, W_az, b_az, W_el, b_el,
                                  out, baseWs);
  k_conv<<<dim3(32, 4, 24), 256, 0, stream>>>(rs, conv1_w, conv1_b,
                                              conv2_w, conv2_b, pooledSum);
  k_learned<<<8, 256, 0, stream>>>(pooledSum, W_res, b_res, learned);
  k_scan<<<32, 256, 0, stream>>>(rs, sconv_w, sconv_b, ctx);
  k_final<<<64, 256, 0, stream>>>(baseWs, learned, ctx, W_proj, b_proj,
                                  cnn_gain, sc_gain, out);
}
